// ViT_88407606821039
// MI455X (gfx1250) — matrix operand order repaired and checked by reading, not compiled, not run on hardware
//
#include <hip/hip_runtime.h>
#include <cstdint>
#include <cstddef>

// ---------------------------------------------------------------------------
// Types for CDNA5 WMMA / TDM
// ---------------------------------------------------------------------------
typedef _Float16 v16h __attribute__((ext_vector_type(16)));
typedef _Float16 v8h  __attribute__((ext_vector_type(8)));
typedef float    v8f  __attribute__((ext_vector_type(8)));
typedef uint32_t u32x4 __attribute__((ext_vector_type(4)));
typedef int32_t  i32x4 __attribute__((ext_vector_type(4)));
typedef int32_t  i32x8 __attribute__((ext_vector_type(8)));

// Problem constants (ViT-Base, N=32)
#define NB     32
#define CCH    3
#define PGRID  14
#define INDIM  768
#define DMODEL 768
#define NHEAD  12
#define DHEAD  64
#define NLAYER 12
#define SEQ    197
#define SPAD   256          // padded seq (multiple of 64 for WMMA N, 128 for K chunks)
#define MLPD   3072
#define OUTD   1000
#define OUTPAD 1024
#define MTOK   (NB*SEQ)     // 6304
#define MPATCH (NB*PGRID*PGRID) // 6272

// Epilogue modes
#define MODE_F32      0   // C = acc*scale + bias   (fp32 store)
#define MODE_ADDF32   1   // C += acc*scale + bias  (fp32 rmw)
#define MODE_F16      2   // C = (f16)(acc*scale + bias)
#define MODE_GELU16   3   // C = (f16)gelu(acc*scale + bias)

// Tensor Data Mover availability (device pass only)
#if defined(__HIP_DEVICE_COMPILE__)
#  if __has_builtin(__builtin_amdgcn_tensor_load_to_lds)
#    define HAVE_TDM 1
#    pragma message("CDNA5 device pass: TDM tensor_load_to_lds path ENABLED")
#  else
#    define HAVE_TDM 0
#    pragma message("CDNA5 device pass: TDM builtin NOT available, using LDS copy fallback")
#  endif
#else
#  define HAVE_TDM 0
#endif

#if HAVE_TDM
// Issue one TDM 2-D tile load (no wait): Bt[64 rows x kc halves] (row stride
// ldbt halves) -> LDS at lds_off, rows contiguous ([64][kc] f16). Wave-level.
__device__ inline void tdm_issue_b_panel(uint32_t lds_off, const _Float16* gsrc,
                                         int kc, int ldbt)
{
    const uint64_t ga = (uint64_t)(uintptr_t)gsrc;
    u32x4 g0;
    g0[0] = 1u;                                   // count=1, user descriptor
    g0[1] = lds_off;                              // lds_addr (bytes)
    g0[2] = (uint32_t)ga;                         // global_addr[31:0]
    g0[3] = (uint32_t)((ga >> 32) & 0x1FFFFFFu)   // global_addr[56:32]
          | (2u << 30);                           // type = 2 ("image")
    i32x8 g1;
    g1[0] = (int)(1u << 16);                      // data_size = 1 (2 bytes)
    g1[1] = (int)(((uint32_t)kc & 0xFFFFu) << 16);           // tensor_dim0[15:0]
    g1[2] = (int)((((uint32_t)kc >> 16) & 0xFFFFu)           // tensor_dim0[31:16]
          | (64u << 16));                                    // tensor_dim1[15:0]
    g1[3] = (int)(((uint32_t)kc & 0xFFFFu) << 16);           // tile_dim0 = kc
    g1[4] = (int)(64u);                                      // tile_dim1 = 64
    g1[5] = (int)((uint32_t)ldbt);                           // dim0_stride[31:0]
    g1[6] = 0;                                               // stride hi bits
    g1[7] = 0;
    i32x4 gz = {0, 0, 0, 0};
#if __clang_major__ >= 23
    i32x8 gz8 = {0, 0, 0, 0, 0, 0, 0, 0};
    __builtin_amdgcn_tensor_load_to_lds(g0, g1, gz, gz, gz8, 0);
#else
    __builtin_amdgcn_tensor_load_to_lds(g0, g1, gz, gz, 0);
#endif
}
#endif

// ---------------------------------------------------------------------------
// Batched WMMA GEMM:  C[M,Ncols] = A[M,K](f16 rm) @ Bt[Ncols,K]^T (f16, B^T)
// Block = 256 threads = 8 waves; block computes a 256-row x 64-col C panel.
// The 64xK B panel is staged through double-buffered LDS in kc-chunks; wave 0
// drives the Tensor Data Mover (issue chunk i+1, s_wait_tensorcnt(1) => chunk
// i landed, barrier) so the DMA overlaps all 8 waves' compute. Each wave does
// a 32x64 tile = 2 A frags x 4 B frags = 8 v_wmma per 32-K step.
// Ncols % 64 == 0; K % kc == 0 (kc = 128 or K).
// ---------------------------------------------------------------------------
__global__ __launch_bounds__(256)
void k_gemm_wmma(const _Float16* __restrict__ A,
                 const _Float16* __restrict__ Bt,
                 const float* __restrict__ bias,
                 void* __restrict__ C,
                 int M, int Ncols, int K,
                 int lda, int ldbt, int ldc,
                 long long strideA, long long strideBt, long long strideC,
                 long long strideBias,
                 int batch, int mode, float scale, int ncols_real)
{
    __shared__ __attribute__((aligned(32))) _Float16 Bpanel[2][64 * 128];

    const int tid  = threadIdx.x;
    const int lane = tid & 31;
    const int wave = tid >> 5;
    const int half = lane >> 4;
    const int l15  = lane & 15;

    const int npan = Ncols >> 6;
    const int mblk = (M + 255) >> 8;
    int rem = blockIdx.x;
    const int b  = rem / (mblk * npan); rem -= b * (mblk * npan);
    const int mb = rem / npan;
    const int nb = rem % npan;

    const int col0 = nb << 6;
    const int rowW = (mb << 8) + (wave << 5);

    const _Float16* Ab  = A  + strideA  * b;
    const _Float16* Btb = Bt + strideBt * b;

    const int kc = (K >= 128 && (K & 127) == 0) ? 128 : K;

    int r0 = rowW + l15;      if (r0 > M - 1) r0 = M - 1;
    int r1 = rowW + 16 + l15; if (r1 > M - 1) r1 = M - 1;
    const _Float16* pa0 = Ab + (size_t)r0 * lda + 16 * half;
    const _Float16* pa1 = Ab + (size_t)r1 * lda + 16 * half;

    v8f zero = {};
    v8f acc[2][4];
#pragma unroll
    for (int rt = 0; rt < 2; ++rt)
#pragma unroll
        for (int c = 0; c < 4; ++c) acc[rt][c] = zero;

#if HAVE_TDM
    if (wave == 0) {
        tdm_issue_b_panel((uint32_t)(uintptr_t)(void*)&Bpanel[0][0],
                          Btb + (size_t)col0 * ldbt, kc, ldbt);
    }
#endif

    int p = 0;
    for (int k0 = 0; k0 < K; k0 += kc) {
#if HAVE_TDM
        if (wave == 0) {
            if (k0 + kc < K) {
                // prefetch next chunk into the other buffer, then wait for
                // the previous one (TDM ops complete in order per wave)
                tdm_issue_b_panel((uint32_t)(uintptr_t)(void*)&Bpanel[1 - p][0],
                                  Btb + (size_t)col0 * ldbt + k0 + kc, kc, ldbt);
                __builtin_amdgcn_s_wait_tensorcnt(1);
            } else {
                __builtin_amdgcn_s_wait_tensorcnt(0);
            }
        }
        __syncthreads();           // Bpanel[p] now valid for all waves
#else
        {
            const int nvec = (64 * kc) >> 3;          // v8h chunks
            const int cpr  = kc >> 3;                 // chunks per row
            for (int i = tid; i < nvec; i += 256) {
                int row = i / cpr, c8 = i % cpr;
                *(v8h*)&Bpanel[p][(size_t)row * kc + c8 * 8] =
                    *(const v8h*)(Btb + (size_t)(col0 + row) * ldbt + k0 + c8 * 8);
            }
            __syncthreads();
        }
#endif
        const _Float16* BpP = &Bpanel[p][0];
        for (int kk = 0; kk < kc; kk += 32) {
            v8h a0lo = *(const v8h*)(pa0);
            v8h a0hi = *(const v8h*)(pa0 + 8);
            v8h a1lo = *(const v8h*)(pa1);
            v8h a1hi = *(const v8h*)(pa1 + 8);
            __builtin_prefetch(pa0 + 64, 0, 1);
            __builtin_prefetch(pa1 + 64, 0, 1);
            v16h a0, a1;
#pragma unroll
            for (int i = 0; i < 8; ++i) {
                a0[i] = a0lo[i]; a0[8 + i] = a0hi[i];
                a1[i] = a1lo[i]; a1[8 + i] = a1hi[i];
            }
            // Load all four B fragments first, then run 8 back-to-back WMMAs
            // (keeps address VALU out of the WMMA hazard shadow).
            v16h bf[4];
#pragma unroll
            for (int c = 0; c < 4; ++c)
                bf[c] = *(const v16h*)&BpP[(size_t)(c * 16 + l15) * kc
                                           + 16 * half + kk];
#pragma unroll
            for (int c = 0; c < 4; ++c) {
                acc[0][c] = __builtin_amdgcn_wmma_f32_16x16x32_f16(
                    false, a0, false, bf[c], (short)0, acc[0][c], false, false);
                acc[1][c] = __builtin_amdgcn_wmma_f32_16x16x32_f16(
                    false, a1, false, bf[c], (short)0, acc[1][c], false, false);
            }
            pa0 += 32; pa1 += 32;
        }
        __syncthreads();   // all waves done reading Bpanel[p]
        p ^= 1;
    }

    // ---- epilogue: D layout — VGPR r = row rowW + rt*16 + r + 8*half,
    //      col = col0 + c*16 + l15 ----
#pragma unroll
    for (int c = 0; c < 4; ++c) {
        const int n = col0 + c * 16 + l15;
        if (n >= ncols_real) continue;
        const float bv = bias ? bias[strideBias * b + n] : 0.0f;
#pragma unroll
        for (int rt = 0; rt < 2; ++rt) {
#pragma unroll
            for (int r = 0; r < 8; ++r) {
                const int row = rowW + rt * 16 + r + 8 * half;
                if (row >= M) continue;
                float v = acc[rt][c][r] * scale + bv;
                const size_t off = (size_t)strideC * b + (size_t)row * ldc + n;
                if (mode == MODE_F32) {
                    ((float*)C)[off] = v;
                } else if (mode == MODE_ADDF32) {
                    ((float*)C)[off] += v;
                } else if (mode == MODE_F16) {
                    ((_Float16*)C)[off] = (_Float16)v;
                } else { // MODE_GELU16 : exact GELU 0.5x(1+erf(x/sqrt(2)))
                    float g = 0.5f * v * (1.0f + erff(v * 0.70710678118654752f));
                    ((_Float16*)C)[off] = (_Float16)g;
                }
            }
        }
    }
}

// ---------------------------------------------------------------------------
// Transpose-convert weights: W[b][K][Nc] (f32) -> Wt[b][NcPad][K] (f16), zero pad
// ---------------------------------------------------------------------------
__global__ void k_transpose_cvt(const float* __restrict__ W, _Float16* __restrict__ Wt,
                                int K, int Nc, int NcPad, long long total)
{
    for (long long idx = blockIdx.x * (long long)blockDim.x + threadIdx.x;
         idx < total; idx += (long long)gridDim.x * blockDim.x) {
        int k = (int)(idx % K);
        long long rem = idx / K;
        int n = (int)(rem % NcPad);
        long long b = rem / NcPad;
        _Float16 v = (_Float16)0.f;
        if (n < Nc) v = (_Float16)W[((size_t)b * K + k) * Nc + n];
        Wt[idx] = v;
    }
}

// images[N,C,224,224] -> Apatch[6272, 768] f16, col = c*256 + iy*16 + ix
__global__ void k_patchify(const float* __restrict__ img, _Float16* __restrict__ Ap, int total)
{
    for (int idx = blockIdx.x * blockDim.x + threadIdx.x; idx < total;
         idx += gridDim.x * blockDim.x) {
        int col = idx % INDIM;
        int m   = idx / INDIM;
        int n   = m / (PGRID * PGRID);
        int p   = m % (PGRID * PGRID);
        int py = p / PGRID, px = p % PGRID;
        int c  = col >> 8;
        int rr = col & 255;
        int iy = rr >> 4, ix = rr & 15;
        float v = img[(((size_t)n * CCH + c) * 224 + (py * 16 + iy)) * 224 + (px * 16 + ix)];
        Ap[idx] = (_Float16)v;
    }
}

// x[n][s][d] = (s==0 ? cls[d] : tok[n*196+s-1][d]) + pos[s][d]
__global__ void k_assemble_x(const float* __restrict__ tok, const float* __restrict__ cls,
                             const float* __restrict__ pos, float* __restrict__ x, int total)
{
    for (int idx = blockIdx.x * blockDim.x + threadIdx.x; idx < total;
         idx += gridDim.x * blockDim.x) {
        int d = idx % DMODEL;
        int s = (idx / DMODEL) % SEQ;
        int n = idx / (DMODEL * SEQ);
        float v = (s == 0) ? cls[d] : tok[((size_t)n * 196 + (s - 1)) * DMODEL + d];
        x[idx] = v + pos[(size_t)s * DMODEL + d];
    }
}

// LayerNorm row kernel: x[row][0:768] -> f16 out
__global__ __launch_bounds__(256)
void k_layernorm(const float* __restrict__ x, const float* __restrict__ g,
                 const float* __restrict__ bta, _Float16* __restrict__ out, int Dm)
{
    __shared__ float red[256];
    const int row = blockIdx.x;
    const float* xr = x + (size_t)row * Dm;
    float s = 0.f;
    for (int i = threadIdx.x; i < Dm; i += blockDim.x) s += xr[i];
    red[threadIdx.x] = s; __syncthreads();
    for (int o = 128; o > 0; o >>= 1) {
        if (threadIdx.x < o) red[threadIdx.x] += red[threadIdx.x + o];
        __syncthreads();
    }
    const float mu = red[0] / Dm; __syncthreads();
    float q = 0.f;
    for (int i = threadIdx.x; i < Dm; i += blockDim.x) { float d = xr[i] - mu; q += d * d; }
    red[threadIdx.x] = q; __syncthreads();
    for (int o = 128; o > 0; o >>= 1) {
        if (threadIdx.x < o) red[threadIdx.x] += red[threadIdx.x + o];
        __syncthreads();
    }
    const float rstd = rsqrtf(red[0] / Dm + 1e-5f);
    for (int i = threadIdx.x; i < Dm; i += blockDim.x)
        out[(size_t)row * Dm + i] = (_Float16)((xr[i] - mu) * rstd * g[i] + bta[i]);
}

// qtmp/ktmp/vtmp [h][6304][64] f16  ->  q16/k16 [n][h][SPAD][64] (zero padded)
//                                    and vT16 [n][h][64][SPAD]
__global__ void k_repack_qkv(const _Float16* __restrict__ qt, const _Float16* __restrict__ kt,
                             const _Float16* __restrict__ vt,
                             _Float16* __restrict__ q16, _Float16* __restrict__ k16,
                             _Float16* __restrict__ vT16, int total)
{
    for (int idx = blockIdx.x * blockDim.x + threadIdx.x; idx < total;
         idx += gridDim.x * blockDim.x) {
        int d   = idx & 63;
        int tmp = idx >> 6;
        int s   = tmp & (SPAD - 1);
        int bh  = tmp / SPAD;
        int h   = bh % NHEAD;
        int n   = bh / NHEAD;
        _Float16 qv = (_Float16)0.f, kv = (_Float16)0.f, vv = (_Float16)0.f;
        if (s < SEQ) {
            size_t src = ((size_t)h * MTOK + (size_t)n * SEQ + s) * DHEAD + d;
            qv = qt[src]; kv = kt[src]; vv = vt[src];
        }
        q16[idx] = qv;
        k16[idx] = kv;
        vT16[((size_t)bh * DHEAD + d) * SPAD + s] = vv;
    }
}

// Softmax over attention rows (in place, f16). Pads cols [SEQ,SPAD) with 0.
__global__ __launch_bounds__(256)
void k_softmax_attn(_Float16* __restrict__ sc)
{
    __shared__ float red[256];
    const int b = blockIdx.x / SEQ;
    const int s = blockIdx.x % SEQ;
    _Float16* row = sc + ((size_t)b * SPAD + s) * SPAD;
    const int tid = threadIdx.x;
    float v = (tid < SEQ) ? (float)row[tid] : -1e30f;
    red[tid] = v; __syncthreads();
    for (int o = 128; o > 0; o >>= 1) {
        if (tid < o) red[tid] = fmaxf(red[tid], red[tid + o]);
        __syncthreads();
    }
    const float mx = red[0]; __syncthreads();
    float e = (tid < SEQ) ? __expf(v - mx) : 0.f;
    red[tid] = e; __syncthreads();
    for (int o = 128; o > 0; o >>= 1) {
        if (tid < o) red[tid] += red[tid + o];
        __syncthreads();
    }
    const float inv = 1.f / red[0];
    if (tid < SPAD) row[tid] = (_Float16)(e * inv);
}

// x[n][s][h*64+d] += o[n][h][s][d]
__global__ void k_add_attn(const float* __restrict__ o, float* __restrict__ x, int total)
{
    for (int idx = blockIdx.x * blockDim.x + threadIdx.x; idx < total;
         idx += gridDim.x * blockDim.x) {
        int col = idx % DMODEL;
        int s   = (idx / DMODEL) % SEQ;
        int n   = idx / (DMODEL * SEQ);
        int h = col >> 6, d = col & 63;
        x[idx] += o[(((size_t)n * NHEAD + h) * SPAD + s) * DHEAD + d];
    }
}

// cls16[n][d] = (f16) x[n][0][d]
__global__ void k_cls(const float* __restrict__ x, _Float16* __restrict__ cls16, int total)
{
    for (int idx = blockIdx.x * blockDim.x + threadIdx.x; idx < total;
         idx += gridDim.x * blockDim.x) {
        int n = idx / DMODEL, d = idx % DMODEL;
        cls16[idx] = (_Float16)x[(size_t)n * SEQ * DMODEL + d];
    }
}

// Final softmax over 1000 logits per image
__global__ __launch_bounds__(256)
void k_softmax_out(const float* __restrict__ logits, float* __restrict__ out)
{
    __shared__ float red[256];
    const int n = blockIdx.x;
    const float* lr = logits + (size_t)n * OUTD;
    float m = -1e30f;
    for (int i = threadIdx.x; i < OUTD; i += 256) m = fmaxf(m, lr[i]);
    red[threadIdx.x] = m; __syncthreads();
    for (int o = 128; o > 0; o >>= 1) {
        if (threadIdx.x < o) red[threadIdx.x] = fmaxf(red[threadIdx.x], red[threadIdx.x + o]);
        __syncthreads();
    }
    const float mx = red[0]; __syncthreads();
    float s = 0.f;
    for (int i = threadIdx.x; i < OUTD; i += 256) s += __expf(lr[i] - mx);
    red[threadIdx.x] = s; __syncthreads();
    for (int o = 128; o > 0; o >>= 1) {
        if (threadIdx.x < o) red[threadIdx.x] += red[threadIdx.x + o];
        __syncthreads();
    }
    const float inv = 1.f / red[0];
    for (int i = threadIdx.x; i < OUTD; i += 256)
        out[(size_t)n * OUTD + i] = __expf(lr[i] - mx) * inv;
}

// ---------------------------------------------------------------------------
// Host side
// ---------------------------------------------------------------------------
static inline int ew_blocks(long long total) {
    long long b = (total + 255) / 256;
    if (b > 4096) b = 4096;
    return (int)b;
}

extern "C" void kernel_launch(void* const* d_in, const int* in_sizes, int n_in,
                              void* d_out, int out_size, void* d_ws, size_t ws_size,
                              hipStream_t stream) {
    (void)in_sizes; (void)n_in; (void)out_size; (void)ws_size;

    const float* images  = (const float*)d_in[0];
    const float* W_map   = (const float*)d_in[1];
    const float* b_map   = (const float*)d_in[2];
    const float* cls_tok = (const float*)d_in[3];
    const float* pos_emb = (const float*)d_in[4];
    const float* ln1_g   = (const float*)d_in[5];
    const float* ln1_b   = (const float*)d_in[6];
    const float* Wq      = (const float*)d_in[7];
    const float* bq      = (const float*)d_in[8];
    const float* Wk      = (const float*)d_in[9];
    const float* bk      = (const float*)d_in[10];
    const float* Wv      = (const float*)d_in[11];
    const float* bv      = (const float*)d_in[12];
    const float* ln2_g   = (const float*)d_in[13];
    const float* ln2_b   = (const float*)d_in[14];
    const float* W1      = (const float*)d_in[15];
    const float* b1      = (const float*)d_in[16];
    const float* W2      = (const float*)d_in[17];
    const float* b2      = (const float*)d_in[18];
    const float* W_head  = (const float*)d_in[19];
    const float* b_head  = (const float*)d_in[20];

    // Workspace carve-out (all offsets 256B aligned)
    size_t off = 0;
    auto alloc = [&](size_t bytes) -> void* {
        void* p = (char*)d_ws + off;
        off += (bytes + 255) & ~(size_t)255;
        return p;
    };
    float*    x      = (float*)   alloc((size_t)MTOK * DMODEL * 4);
    _Float16* Ap     = (_Float16*)alloc((size_t)MPATCH * INDIM * 2);
    float*    tok    = (float*)   alloc((size_t)MPATCH * DMODEL * 4);
    _Float16* h16    = (_Float16*)alloc((size_t)MTOK * DMODEL * 2);
    _Float16* qtmp   = (_Float16*)alloc((size_t)NHEAD * MTOK * DHEAD * 2);
    _Float16* ktmp   = (_Float16*)alloc((size_t)NHEAD * MTOK * DHEAD * 2);
    _Float16* vtmp   = (_Float16*)alloc((size_t)NHEAD * MTOK * DHEAD * 2);
    _Float16* q16    = (_Float16*)alloc((size_t)NB * NHEAD * SPAD * DHEAD * 2);
    _Float16* k16    = (_Float16*)alloc((size_t)NB * NHEAD * SPAD * DHEAD * 2);
    _Float16* vT16   = (_Float16*)alloc((size_t)NB * NHEAD * DHEAD * SPAD * 2);
    float*    obuf   = (float*)   alloc((size_t)NB * NHEAD * SPAD * DHEAD * 4);
    _Float16* sc     = (_Float16*)alloc((size_t)NB * NHEAD * SPAD * SPAD * 2);
    _Float16* mid16  = (_Float16*)alloc((size_t)MTOK * MLPD * 2);
    _Float16* Wmap_t = (_Float16*)alloc((size_t)DMODEL * INDIM * 2);
    _Float16* Wqt    = (_Float16*)alloc((size_t)NHEAD * DHEAD * DHEAD * 2);
    _Float16* Wkt    = (_Float16*)alloc((size_t)NHEAD * DHEAD * DHEAD * 2);
    _Float16* Wvt    = (_Float16*)alloc((size_t)NHEAD * DHEAD * DHEAD * 2);
    _Float16* W1t    = (_Float16*)alloc((size_t)MLPD * DMODEL * 2);
    _Float16* W2t    = (_Float16*)alloc((size_t)DMODEL * MLPD * 2);
    _Float16* Wht    = (_Float16*)alloc((size_t)OUTPAD * DMODEL * 2);
    _Float16* cls16  = (_Float16*)alloc((size_t)NB * DMODEL * 2);
    float*    logits = (float*)   alloc((size_t)NB * OUTD * 4);

    auto gemm = [&](const _Float16* A, const _Float16* Bt, const float* bias, void* C,
                    int M, int Nc, int K, int lda, int ldbt, int ldc,
                    long long sA, long long sB, long long sC, long long sBias,
                    int batch, int mode, float scale, int ncr) {
        int mblk = (M + 255) / 256;
        int npan = Nc / 64;
        int blocks = batch * mblk * npan;
        k_gemm_wmma<<<blocks, 256, 0, stream>>>(A, Bt, bias, C, M, Nc, K,
                                                lda, ldbt, ldc, sA, sB, sC, sBias,
                                                batch, mode, scale, ncr);
    };

    // ---- Patch embedding --------------------------------------------------
    {
        long long tot = (long long)DMODEL * INDIM;
        k_transpose_cvt<<<ew_blocks(tot), 256, 0, stream>>>(W_map, Wmap_t,
                                                            INDIM, DMODEL, DMODEL, tot);
        k_patchify<<<ew_blocks((long long)MPATCH * INDIM), 256, 0, stream>>>(
            images, Ap, MPATCH * INDIM);
        gemm(Ap, Wmap_t, b_map, tok, MPATCH, DMODEL, INDIM, INDIM, INDIM, DMODEL,
             0, 0, 0, 0, 1, MODE_F32, 1.f, DMODEL);
        k_assemble_x<<<ew_blocks((long long)MTOK * DMODEL), 256, 0, stream>>>(
            tok, cls_tok, pos_emb, x, MTOK * DMODEL);
    }

    // ---- Transformer blocks ----------------------------------------------
    for (int l = 0; l < NLAYER; ++l) {
        // MHSA
        k_layernorm<<<MTOK, 256, 0, stream>>>(x, ln1_g + (size_t)l * DMODEL,
                                              ln1_b + (size_t)l * DMODEL, h16, DMODEL);
        const long long wqkv = (long long)NHEAD * DHEAD * DHEAD;
        k_transpose_cvt<<<ew_blocks(wqkv), 256, 0, stream>>>(
            Wq + (size_t)l * wqkv, Wqt, DHEAD, DHEAD, DHEAD, wqkv);
        k_transpose_cvt<<<ew_blocks(wqkv), 256, 0, stream>>>(
            Wk + (size_t)l * wqkv, Wkt, DHEAD, DHEAD, DHEAD, wqkv);
        k_transpose_cvt<<<ew_blocks(wqkv), 256, 0, stream>>>(
            Wv + (size_t)l * wqkv, Wvt, DHEAD, DHEAD, DHEAD, wqkv);
        // per-head projections: A = h16 column window, batch over heads
        gemm(h16, Wqt, bq + (size_t)l * NHEAD * DHEAD, qtmp, MTOK, DHEAD, DHEAD,
             DMODEL, DHEAD, DHEAD, DHEAD, (long long)DHEAD * DHEAD,
             (long long)MTOK * DHEAD, DHEAD, NHEAD, MODE_F16, 1.f, DHEAD);
        gemm(h16, Wkt, bk + (size_t)l * NHEAD * DHEAD, ktmp, MTOK, DHEAD, DHEAD,
             DMODEL, DHEAD, DHEAD, DHEAD, (long long)DHEAD * DHEAD,
             (long long)MTOK * DHEAD, DHEAD, NHEAD, MODE_F16, 1.f, DHEAD);
        gemm(h16, Wvt, bv + (size_t)l * NHEAD * DHEAD, vtmp, MTOK, DHEAD, DHEAD,
             DMODEL, DHEAD, DHEAD, DHEAD, (long long)DHEAD * DHEAD,
             (long long)MTOK * DHEAD, DHEAD, NHEAD, MODE_F16, 1.f, DHEAD);
        k_repack_qkv<<<ew_blocks((long long)NB * NHEAD * SPAD * DHEAD), 256, 0, stream>>>(
            qtmp, ktmp, vtmp, q16, k16, vT16, NB * NHEAD * SPAD * DHEAD);
        // scores = Q @ K^T * (1/8); Bt form of K^T is K itself
        gemm(q16, k16, nullptr, sc, SPAD, SPAD, DHEAD, DHEAD, DHEAD, SPAD,
             (long long)SPAD * DHEAD, (long long)SPAD * DHEAD,
             (long long)SPAD * SPAD, 0, NB * NHEAD, MODE_F16, 0.125f, SPAD);
        k_softmax_attn<<<NB * NHEAD * SEQ, 256, 0, stream>>>(sc);
        // O = attn @ V; Bt form of V is V^T
        gemm(sc, vT16, nullptr, obuf, SPAD, DHEAD, SPAD, SPAD, SPAD, DHEAD,
             (long long)SPAD * SPAD, (long long)DHEAD * SPAD,
             (long long)SPAD * DHEAD, 0, NB * NHEAD, MODE_F32, 1.f, DHEAD);
        k_add_attn<<<ew_blocks((long long)MTOK * DMODEL), 256, 0, stream>>>(
            obuf, x, MTOK * DMODEL);

        // MLP
        k_layernorm<<<MTOK, 256, 0, stream>>>(x, ln2_g + (size_t)l * DMODEL,
                                              ln2_b + (size_t)l * DMODEL, h16, DMODEL);
        {
            long long tot1 = (long long)MLPD * DMODEL;
            k_transpose_cvt<<<ew_blocks(tot1), 256, 0, stream>>>(
                W1 + (size_t)l * DMODEL * MLPD, W1t, DMODEL, MLPD, MLPD, tot1);
            gemm(h16, W1t, b1 + (size_t)l * MLPD, mid16, MTOK, MLPD, DMODEL,
                 DMODEL, DMODEL, MLPD, 0, 0, 0, 0, 1, MODE_GELU16, 1.f, MLPD);
            long long tot2 = (long long)DMODEL * MLPD;
            k_transpose_cvt<<<ew_blocks(tot2), 256, 0, stream>>>(
                W2 + (size_t)l * MLPD * DMODEL, W2t, MLPD, DMODEL, DMODEL, tot2);
            gemm(mid16, W2t, b2 + (size_t)l * DMODEL, x, MTOK, DMODEL, MLPD,
                 MLPD, MLPD, DMODEL, 0, 0, 0, 0, 1, MODE_ADDF32, 1.f, DMODEL);
        }
    }

    // ---- Head + softmax ---------------------------------------------------
    {
        long long tot = (long long)OUTPAD * DMODEL;
        k_transpose_cvt<<<ew_blocks(tot), 256, 0, stream>>>(W_head, Wht,
                                                            DMODEL, OUTD, OUTPAD, tot);
        k_cls<<<ew_blocks((long long)NB * DMODEL), 256, 0, stream>>>(x, cls16, NB * DMODEL);
        gemm(cls16, Wht, b_head, logits, NB, OUTPAD, DMODEL, DMODEL, DMODEL, OUTD,
             0, 0, 0, 0, 1, MODE_F32, 1.f, OUTD);
        k_softmax_out<<<NB, 256, 0, stream>>>(logits, (float*)d_out);
    }
}
